// RoIHead_77910706749628
// MI455X (gfx1250) — matrix operand order repaired and checked by reading, not compiled, not run on hardware
//
#include <hip/hip_runtime.h>

typedef __attribute__((ext_vector_type(16))) __bf16 v16bf;
typedef __attribute__((ext_vector_type(8)))  float  v8f;
typedef __attribute__((ext_vector_type(4)))  float  f4;

union ABv { v16bf v; f4 q[2]; };

#define NROI   512
#define CCH    512
#define HH     50
#define WW     50
#define PP     7
#define FEATD  25088   // CCH*PP*PP
#define HID    4096
#define NCLS   21
#define MHEAD  105     // 84 bbox + 21 cls

#define KTILE   64
#define ASTRIDE 72     // KTILE + 8 pad (bf16 units) -> 144B row stride
#define BSTRIDE 72

// ---------------------------------------------------------------------------
// RoIPool: one block per roi, 256 threads cover 25088 (c,ph,pw) outputs.
// Writes the A-matrix of FC1 directly in bf16, row-major [NROI, FEATD].
// ---------------------------------------------------------------------------
__global__ __launch_bounds__(256) void roipool_kernel(
    const float* __restrict__ x, const float* __restrict__ rois,
    const int* __restrict__ ridx, __bf16* __restrict__ A)
{
    const int n   = blockIdx.x;
    const int tid = threadIdx.x;
    const float SCALE = 1.0f / 16.0f;

    // reference swaps: boxes = rois[:, [1,0,3,2]]
    const float y1 = rois[n * 4 + 0];
    const float x1 = rois[n * 4 + 1];
    const float y2 = rois[n * 4 + 2];
    const float x2 = rois[n * 4 + 3];
    const float xs = roundf(x1 * SCALE);
    const float ys = roundf(y1 * SCALE);
    const float xe = roundf(x2 * SCALE);
    const float ye = roundf(y2 * SCALE);
    const float roi_w = fmaxf(xe - xs + 1.0f, 1.0f);
    const float roi_h = fmaxf(ye - ys + 1.0f, 1.0f);
    const float bw = roi_w / (float)PP;
    const float bh = roi_h / (float)PP;

    const int b = ridx[n];
    const float* __restrict__ feat = x + (size_t)b * CCH * HH * WW;
    __bf16* __restrict__ arow = A + (size_t)n * FEATD;

    for (int col = tid; col < FEATD; col += 256) {
        const int c   = col / (PP * PP);
        const int rem = col % (PP * PP);
        const int ph  = rem / PP;
        const int pw  = rem % PP;

        float wsf = floorf((float)pw * bw) + xs;
        float wef = ceilf(((float)pw + 1.0f) * bw) + xs;
        float hsf = floorf((float)ph * bh) + ys;
        float hef = ceilf(((float)ph + 1.0f) * bh) + ys;
        int ws_ = (int)fminf(fmaxf(wsf, 0.0f), (float)WW);
        int we_ = (int)fminf(fmaxf(wef, 0.0f), (float)WW);
        int hs_ = (int)fminf(fmaxf(hsf, 0.0f), (float)HH);
        int he_ = (int)fminf(fmaxf(hef, 0.0f), (float)HH);

        float mx = -3.402823466e38f;
        const float* __restrict__ fc = feat + (size_t)c * HH * WW;
        for (int h = hs_; h < he_; ++h) {
            const float* __restrict__ fr = fc + h * WW;
            for (int w = ws_; w < we_; ++w)
                mx = fmaxf(mx, fr[w]);
        }
        float out = (hs_ >= he_ || ws_ >= we_) ? 0.0f : mx;
        arow[col] = (__bf16)out;
    }
}

// ---------------------------------------------------------------------------
// Tiled bf16 WMMA GEMM: Out[N, M] = relu(A[N, K] * W[K, M] + bias)
//   A : bf16 row-major, async global->LDS b128, DOUBLE BUFFERED
//   W : f32 row-major, non-temporal b128 loads -> bf16 -> LDS transposed
// Block = 256 thr (8 waves), tile = 128 rows x 64 cols, K staged 64 wide.
// Each wave: 2x2 tiles of v_wmma_f32_16x16x32_bf16 per K=32 step.
// ---------------------------------------------------------------------------
__global__ __launch_bounds__(256) void gemm_bf16_kernel(
    const __bf16* __restrict__ A, const float* __restrict__ W,
    const float* __restrict__ bias, __bf16* __restrict__ Out,
    int Kdim, int M)
{
    __shared__ __align__(16) __bf16 Alds[2][128 * ASTRIDE];
    __shared__ __align__(16) __bf16 Blds[2][64 * BSTRIDE];

    const int tid  = threadIdx.x;
    const int lane = tid & 31;
    const int wv   = tid >> 5;          // 0..7
    const int wr   = wv >> 1;           // 0..3  row block of wave
    const int wc   = wv & 1;            // 0..1  col block of wave
    const int l    = lane & 15;
    const int hi   = lane >> 4;

    const int rowBase = blockIdx.y * 128;
    const int colBase = blockIdx.x * 64;

    // per-thread staging coordinates (constant across stages)
    const int a_row0 = tid >> 3;        // A chunk row, + i*32
    const int a_kc   = (tid & 7) * 8;   // A bf16 offset within row
    const int b_kr0  = tid >> 4;        // B K-row, + i*16
    const int b_mc   = (tid & 15) * 4;  // B col group of 4

    const unsigned long long Ablk =
        (unsigned long long)(A + (size_t)rowBase * Kdim);

    // issue 4 async 16B global->LDS copies for one A stage
    auto issueA = [&](int kb, int buf) {
        #pragma unroll
        for (int i = 0; i < 4; ++i) {
            const int row = a_row0 + i * 32;
            unsigned goff = ((unsigned)row * (unsigned)Kdim +
                             (unsigned)(kb + a_kc)) * 2u;
            unsigned loff =
                (unsigned)(size_t)(void*)&Alds[buf][row * ASTRIDE + a_kc];
            asm volatile("global_load_async_to_lds_b128 %0, %1, %2"
                         :: "v"(loff), "v"(goff), "s"(Ablk) : "memory");
        }
    };
    // load one B stage (64x64 f32) into registers, non-temporal (streamed once)
    auto loadB = [&](int kb, f4* r) {
        #pragma unroll
        for (int i = 0; i < 4; ++i)
            r[i] = __builtin_nontemporal_load(
                (const f4*)(W + (size_t)(kb + b_kr0 + i * 16) * M +
                            colBase + b_mc));
    };
    // convert + store one B stage transposed [m][k] into LDS
    auto storeB = [&](int buf, const f4* r) {
        #pragma unroll
        for (int i = 0; i < 4; ++i) {
            const int kr = b_kr0 + i * 16;
            Blds[buf][(b_mc + 0) * BSTRIDE + kr] = (__bf16)r[i].x;
            Blds[buf][(b_mc + 1) * BSTRIDE + kr] = (__bf16)r[i].y;
            Blds[buf][(b_mc + 2) * BSTRIDE + kr] = (__bf16)r[i].z;
            Blds[buf][(b_mc + 3) * BSTRIDE + kr] = (__bf16)r[i].w;
        }
    };

    v8f acc[2][2] = {};
    f4  breg[4];

    issueA(0, 0);
    loadB(0, breg);

    const int S = Kdim / KTILE;
    for (int s = 0; s < S; ++s) {
        const int cur = s & 1;
        const bool hasNext = (s + 1 < S);

        if (hasNext) issueA((s + 1) * KTILE, cur ^ 1);  // prefetch A (async)
        storeB(cur, breg);                              // stage current B
        if (hasNext) loadB((s + 1) * KTILE, breg);      // prefetch B (regs)

        // in-order async completion: 4 newest may stay in flight
        if (hasNext) asm volatile("s_wait_asynccnt 4" ::: "memory");
        else         asm volatile("s_wait_asynccnt 0" ::: "memory");
        __syncthreads();

        const __bf16* __restrict__ Ab = Alds[cur];
        const __bf16* __restrict__ Bb = Blds[cur];
        #pragma unroll
        for (int ks = 0; ks < 2; ++ks) {
            const int kk = ks * 32;
            #pragma unroll
            for (int rt = 0; rt < 2; ++rt) {
                // A 16x32 bf16: lane l = row, K = {hi*8+0..7, 16+hi*8+0..7}
                ABv a;
                const __bf16* ap =
                    &Ab[(wr * 32 + rt * 16 + l) * ASTRIDE + kk + hi * 8];
                a.q[0] = *(const f4*)ap;
                a.q[1] = *(const f4*)(ap + 16);
                #pragma unroll
                for (int ct = 0; ct < 2; ++ct) {
                    // B 32x16 bf16: lane l = col, K = {hi*8+0..7, 16+hi*8+0..7}
                    ABv bq;
                    const __bf16* bp =
                        &Bb[(wc * 32 + ct * 16 + l) * BSTRIDE + kk + hi * 8];
                    bq.q[0] = *(const f4*)bp;
                    bq.q[1] = *(const f4*)(bp + 16);
                    acc[rt][ct] = __builtin_amdgcn_wmma_f32_16x16x32_bf16(
                        false, a.v, false, bq.v, (short)0, acc[rt][ct],
                        false, false);
                }
            }
        }
        __syncthreads();
    }

    // --- epilogue: bias + ReLU + bf16 store
    #pragma unroll
    for (int rt = 0; rt < 2; ++rt) {
        #pragma unroll
        for (int ct = 0; ct < 2; ++ct) {
            const int gcol = colBase + wc * 32 + ct * 16 + l;
            const float bv = bias[gcol];
            const int growBase = rowBase + wr * 32 + rt * 16 + hi * 8;
            #pragma unroll
            for (int j = 0; j < 8; ++j) {
                float v = acc[rt][ct][j] + bv;
                v = v > 0.0f ? v : 0.0f;
                Out[(size_t)(growBase + j) * M + gcol] = (__bf16)v;
            }
        }
    }
}

// ---------------------------------------------------------------------------
// Heads: one block per roi; 105 lanes each do a 4096-dot against Wb/Wc.
// Hidden row cached in LDS (bf16). Tiny (<0.5% of total FLOPs).
// ---------------------------------------------------------------------------
__global__ __launch_bounds__(128) void heads_kernel(
    const __bf16* __restrict__ H, const float* __restrict__ Wb,
    const float* __restrict__ bb, const float* __restrict__ Wc,
    const float* __restrict__ bc, float* __restrict__ out)
{
    __shared__ __align__(16) __bf16 hrow[HID];
    const int n = blockIdx.x, tid = threadIdx.x;

    const uint4* __restrict__ src = (const uint4*)(H + (size_t)n * HID);
    uint4* dst = (uint4*)hrow;
    #pragma unroll
    for (int i = 0; i < 4; ++i) dst[tid + i * 128] = src[tid + i * 128];
    __syncthreads();

    if (tid < MHEAD) {
        const bool isb = tid < 84;
        const int mm = isb ? tid : tid - 84;
        const int ld = isb ? 84 : NCLS;
        const float* __restrict__ Wp = isb ? Wb : Wc;
        float acc = isb ? bb[mm] : bc[mm];
        #pragma unroll 8
        for (int k = 0; k < HID; ++k)
            acc += (float)hrow[k] * Wp[(size_t)k * ld + mm];
        if (isb) out[(size_t)n * 84 + tid] = acc;
        else     out[(size_t)NROI * 84 + (size_t)n * NCLS + mm] = acc;
    }
}

// ---------------------------------------------------------------------------
extern "C" void kernel_launch(void* const* d_in, const int* in_sizes, int n_in,
                              void* d_out, int out_size, void* d_ws, size_t ws_size,
                              hipStream_t stream) {
    const float* x    = (const float*)d_in[0];
    const float* rois = (const float*)d_in[1];
    const int*   ridx = (const int*)  d_in[2];
    const float* W1   = (const float*)d_in[3];
    const float* b1   = (const float*)d_in[4];
    const float* W2   = (const float*)d_in[5];
    const float* b2   = (const float*)d_in[6];
    const float* Wb   = (const float*)d_in[7];
    const float* bb   = (const float*)d_in[8];
    const float* Wc   = (const float*)d_in[9];
    const float* bc   = (const float*)d_in[10];
    float* out = (float*)d_out;

    __bf16* A1 = (__bf16*)d_ws;                      // [512, 25088] bf16
    __bf16* H1 = A1 + (size_t)NROI * FEATD;          // [512, 4096]  bf16
    __bf16* H2 = H1 + (size_t)NROI * HID;            // [512, 4096]  bf16

    roipool_kernel<<<NROI, 256, 0, stream>>>(x, rois, ridx, A1);

    dim3 g1(HID / 64, NROI / 128);                   // (64, 4)
    gemm_bf16_kernel<<<g1, 256, 0, stream>>>(A1, W1, b1, H1, FEATD, HID);
    gemm_bf16_kernel<<<g1, 256, 0, stream>>>(H1, W2, b2, H2, HID, HID);

    heads_kernel<<<NROI, 128, 0, stream>>>(H2, Wb, bb, Wc, bc, out);
}
